// MultiheadAttention_56160992363016
// MI455X (gfx1250) — matrix operand order repaired and checked by reading, not compiled, not run on hardware
//
#include <hip/hip_runtime.h>

// ---------------------------------------------------------------------------
// MultiheadAttention forward for MI455X (gfx1250, wave32, WMMA bf16,
// async global->LDS staging of block-shared tiles).
// L=S=2048, N=4, E=1024, H=16, D=64
// ---------------------------------------------------------------------------

typedef __bf16 bf16;
typedef __attribute__((ext_vector_type(16))) __bf16 v16bf;
typedef __attribute__((ext_vector_type(8)))  __bf16 v8bf;
typedef __attribute__((ext_vector_type(4)))  __bf16 v4bf;
typedef __attribute__((ext_vector_type(8)))  float  v8f;
typedef __attribute__((ext_vector_type(4)))  float  v4f;

constexpr int L_  = 2048;
constexpr int S_  = 2048;
constexpr int NB_ = 4;
constexpr int E_  = 1024;
constexpr int H_  = 16;
constexpr int D_  = 64;

constexpr int KP_ = 72;  // padded LDS row stride (elems) for 64-wide K rows
constexpr int WP_ = 40;  // padded LDS row stride (elems) for 32-wide rows

// ---------------------------------------------------------------------------
// CDNA5 async copy: 16 bytes global -> LDS per lane (ASYNCcnt-tracked).
// LDS address = low 32 bits of the generic shared pointer.
// ---------------------------------------------------------------------------
__device__ inline void async_cp16(void* ldst, const void* gsrc) {
  asm volatile("global_load_async_to_lds_b128 %0, %1, off"
               :: "v"((unsigned)(unsigned long long)ldst), "v"(gsrc)
               : "memory");
}

__device__ inline void wait_async0() {
#if __has_builtin(__builtin_amdgcn_s_wait_asynccnt)
  __builtin_amdgcn_s_wait_asynccnt(0);
#else
  asm volatile("s_wait_asynccnt 0" ::: "memory");
#endif
}

// ---------------------------------------------------------------------------
// WMMA helper: D = A(16x32 bf16) * B(32x16 bf16) + C(16x16 f32)
// ---------------------------------------------------------------------------
__device__ inline v8f wmma_bf16(v16bf a, v16bf b, v8f c) {
  return __builtin_amdgcn_wmma_f32_16x16x32_bf16(
      /*neg_a=*/false, a, /*neg_b=*/false, b,
      /*c_mod=*/(short)0, c, /*reuse_a=*/false, /*reuse_b=*/false);
}

// A-fragment (16x32). lane<16: row=lane, K=k0..k0+15;
// lane>=16: same row, K=k0+16..k0+31 (contiguous, matches B fragments).
__device__ inline v16bf a_frag_f32(const float* rowp, int k0, int hi) {
  const float* p = rowp + k0 + (hi ? 16 : 0);
  v4f x0 = *(const v4f*)(p + 0);
  v4f x1 = *(const v4f*)(p + 4);
  v4f x2 = *(const v4f*)(p + 8);
  v4f x3 = *(const v4f*)(p + 12);
  v16bf a;
#pragma unroll
  for (int i = 0; i < 4; ++i) {
    a[i]      = (__bf16)x0[i];
    a[i + 4]  = (__bf16)x1[i];
    a[i + 8]  = (__bf16)x2[i];
    a[i + 12] = (__bf16)x3[i];
  }
  return a;
}

__device__ inline v16bf a_frag_bf16(const bf16* rowp, int k0, int hi) {
  const bf16* p = rowp + k0 + (hi ? 16 : 0);
  v8bf lo = *(const v8bf*)(p);
  v8bf hh = *(const v8bf*)(p + 8);
  v16bf a;
#pragma unroll
  for (int i = 0; i < 8; ++i) { a[i] = lo[i]; a[i + 8] = hh[i]; }
  return a;
}

// B-fragment read (32x16) from an LDS panel with row stride `stride` elems:
// lane<16: col=lane, K=0..15; lane>=16: K=16..31 (contiguous within the row).
__device__ inline v16bf b_frag_lds(const bf16* colrow, int hi) {
  return *(const v16bf*)(colrow + hi * 16);
}

// ---------------------------------------------------------------------------
// Kernel 0: convert projection weights fp32 -> bf16 (read many times later).
// ---------------------------------------------------------------------------
__global__ __launch_bounds__(256) void convert_w_kernel(
    const float* __restrict__ ipw, const float* __restrict__ opw,
    bf16* __restrict__ ipw_bf, bf16* __restrict__ opw_bf) {
  constexpr size_t NIP = (size_t)3 * E_ * E_;
  constexpr size_t NOP = (size_t)E_ * E_;
  const size_t i = ((size_t)blockIdx.x * blockDim.x + threadIdx.x) * 4;
  if (i < NIP) {
    v4f x = *(const v4f*)(ipw + i);
    v4bf y;
#pragma unroll
    for (int t = 0; t < 4; ++t) y[t] = (__bf16)x[t];
    *(v4bf*)(ipw_bf + i) = y;
  } else {
    const size_t j = i - NIP;
    if (j < NOP) {
      v4f x = *(const v4f*)(opw + j);
      v4bf y;
#pragma unroll
      for (int t = 0; t < 4; ++t) y[t] = (__bf16)x[t];
      *(v4bf*)(opw_bf + j) = y;
    }
  }
}

// ---------------------------------------------------------------------------
// Async stage: 64-column weight panel (64 x 32K bf16) into LDS (padded WP_).
// Each of the 4 waves copies 16 columns (2 async b128 instructions).
// ---------------------------------------------------------------------------
__device__ inline void stage_w(bf16* buf, const bf16* wbase, int k0, int w,
                               int lane) {
#pragma unroll
  for (int j = 0; j < 2; ++j) {
    const int col = w * 16 + j * 8 + (lane >> 2);
    const int ch  = (lane & 3) * 8;  // element offset within the 32-K row
    async_cp16(buf + col * WP_ + ch, wbase + (size_t)col * E_ + k0 + ch);
  }
}

// ---------------------------------------------------------------------------
// Kernel 1: QKV projection. y[t,n,f] = sum_e x[t,n,e]*w[f,e] + b[f]
// Block = 4 waves sharing one 64-col weight panel (async double-buffered in
// LDS). Per wave: 16 rows x 64 cols. q scaled by 1/sqrt(D); v transposed.
// ---------------------------------------------------------------------------
__global__ __launch_bounds__(128) void qkv_proj_kernel(
    const float* __restrict__ query, const float* __restrict__ keyx,
    const float* __restrict__ value, const bf16* __restrict__ wbf,
    const float* __restrict__ ipb,
    bf16* __restrict__ qh, bf16* __restrict__ kh, bf16* __restrict__ vT) {
  __shared__ __align__(16) bf16 bsh[2][64 * WP_];
  const int lane = threadIdx.x & 31;
  const int w    = threadIdx.x >> 5;
  const int hi   = (lane >> 4) & 1;
  const int cl   = lane & 15;
  const int wid  = blockIdx.x * 4 + w;
  const int mt   = wid & 511;       // 512 tiles of 16 rows (8192 rows)
  const int fc   = wid >> 9;        // 48 chunks of 64 cols
  const int which = fc >> 4;        // 0=q, 1=k, 2=v
  const int m0 = mt << 4;
  const int f0 = fc << 6;

  const float* X = (which == 0) ? query : (which == 1) ? keyx : value;
  const float* xrow  = X + (size_t)(m0 + cl) * E_;
  const bf16* wpanel = wbf + (size_t)f0 * E_;

  stage_w(&bsh[0][0], wpanel, 0, w, lane);
  wait_async0();
  __syncthreads();

  v8f acc[4] = {};
  v16bf a_cur = a_frag_f32(xrow, 0, hi);
  for (int it = 0; it < E_ / 32; ++it) {
    const int k0  = it * 32;
    const int cur = it & 1;
    if (k0 + 32 < E_) stage_w(&bsh[cur ^ 1][0], wpanel, k0 + 32, w, lane);
    v16bf a_nxt = (k0 + 32 < E_) ? a_frag_f32(xrow, k0 + 32, hi) : a_cur;
    const bf16* bp = &bsh[cur][0];
    v16bf b0 = b_frag_lds(bp + (size_t)(0  + cl) * WP_, hi);
    v16bf b1 = b_frag_lds(bp + (size_t)(16 + cl) * WP_, hi);
    v16bf b2 = b_frag_lds(bp + (size_t)(32 + cl) * WP_, hi);
    v16bf b3 = b_frag_lds(bp + (size_t)(48 + cl) * WP_, hi);
    acc[0] = wmma_bf16(a_cur, b0, acc[0]);
    acc[1] = wmma_bf16(a_cur, b1, acc[1]);
    acc[2] = wmma_bf16(a_cur, b2, acc[2]);
    acc[3] = wmma_bf16(a_cur, b3, acc[3]);
    a_cur = a_nxt;
    wait_async0();
    __syncthreads();
  }

  const float scale = (which == 0) ? 0.125f : 1.0f;  // 1/sqrt(64)
#pragma unroll
  for (int c = 0; c < 4; ++c) {
    const int f  = f0 + 16 * c + cl;
    const float bias = ipb[f];
    const int fe = f & (E_ - 1);
    const int h  = fe >> 6;
    const int d  = fe & 63;
#pragma unroll
    for (int j = 0; j < 8; ++j) {
      const int r = m0 + j + hi * 8;     // flattened (t*N + n)
      const int t = r >> 2;
      const int n = r & 3;
      const bf16 val = (bf16)((acc[c][j] + bias) * scale);
      const size_t nh = (size_t)(n * H_ + h);
      if (which == 0)      qh[(nh * L_ + t) * D_ + d] = val;
      else if (which == 1) kh[(nh * S_ + t) * D_ + d] = val;
      else                 vT[(nh * D_ + d) * S_ + t] = val;  // transposed
    }
  }
}

// ---------------------------------------------------------------------------
// Async stage of one attention s-tile: K (32 rows x 64d) and V^T (64d x 32s)
// shared by all 4 waves of the block. 4 async b128 instructions per wave.
// ---------------------------------------------------------------------------
__device__ inline void stage_kv(bf16* kbuf, bf16* vbuf, const bf16* kb,
                                const bf16* vb, int s0, int w, int lane) {
#pragma unroll
  for (int j = 0; j < 2; ++j) {  // K tile rows
    const int row = w * 8 + j * 4 + (lane >> 3);
    const int ch  = (lane & 7) * 8;
    async_cp16(kbuf + row * KP_ + ch, kb + (size_t)(s0 + row) * D_ + ch);
  }
#pragma unroll
  for (int j = 0; j < 2; ++j) {  // V^T tile rows (d-major)
    const int dr = w * 16 + j * 8 + (lane >> 2);
    const int ch = (lane & 3) * 8;
    async_cp16(vbuf + dr * WP_ + ch, vb + (size_t)dr * S_ + s0 + ch);
  }
}

// ---------------------------------------------------------------------------
// Kernel 2: flash attention per (n,h,16-row tile); online softmax over S.
// Blocks of 4 waves share (n,h): K/V s-tiles async-staged to LDS, dbl-buffer.
// ---------------------------------------------------------------------------
__global__ __launch_bounds__(128) void flash_attn_kernel(
    const bf16* __restrict__ qh, const bf16* __restrict__ kh,
    const bf16* __restrict__ vT, bf16* __restrict__ ctxb,
    float* __restrict__ mstat, float* __restrict__ lstat) {
  __shared__ __align__(16) bf16 ksh[2][32 * KP_];
  __shared__ __align__(16) bf16 vsh[2][64 * WP_];
  __shared__ __align__(32) bf16 pb[4][16 * 32];  // per-wave P re-stripe
  const int lane = threadIdx.x & 31;
  const int w    = threadIdx.x >> 5;
  const int hi   = (lane >> 4) & 1;
  const int cl   = lane & 15;
  const int wid  = blockIdx.x * 4 + w;
  const int lt   = wid & 127;
  const int nh   = wid >> 7;          // same for all 4 waves of a block
  const int n    = nh >> 4, h = nh & 15;
  const int l0   = lt << 4;

  const bf16* qb = qh + (size_t)nh * L_ * D_;
  const bf16* kb = kh + (size_t)nh * S_ * D_;
  const bf16* vb = vT + (size_t)nh * D_ * S_;

  const bf16* qrow = qb + (size_t)(l0 + cl) * D_;
  const v16bf qA0 = a_frag_bf16(qrow, 0, hi);
  const v16bf qA1 = a_frag_bf16(qrow, 32, hi);

  float mrun[8], lrun[8];
  v8f ctx[4] = {};
#pragma unroll
  for (int j = 0; j < 8; ++j) { mrun[j] = -1e30f; lrun[j] = 0.0f; }

  bf16* pw = &pb[w][0];

  stage_kv(&ksh[0][0], &vsh[0][0], kb, vb, 0, w, lane);
  wait_async0();
  __syncthreads();

  for (int it = 0; it < S_ / 32; ++it) {
    const int s0  = it * 32;
    const int cur = it & 1;
    if (s0 + 32 < S_)
      stage_kv(&ksh[cur ^ 1][0], &vsh[cur ^ 1][0], kb, vb, s0 + 32, w, lane);

    // K fragments from LDS (B layout: column = s index, K = d contiguous)
    const bf16* kc = &ksh[cur][0];
    v16bf kb00 = *(const v16bf*)(kc + (0 * 16 + cl) * KP_ + 0  + hi * 16);
    v16bf kb01 = *(const v16bf*)(kc + (0 * 16 + cl) * KP_ + 32 + hi * 16);
    v16bf kb10 = *(const v16bf*)(kc + (1 * 16 + cl) * KP_ + 0  + hi * 16);
    v16bf kb11 = *(const v16bf*)(kc + (1 * 16 + cl) * KP_ + 32 + hi * 16);

    v8f z = {};
    v8f st0 = wmma_bf16(qA0, kb00, z);
    st0     = wmma_bf16(qA1, kb01, st0);
    v8f st1 = wmma_bf16(qA0, kb10, z);
    st1     = wmma_bf16(qA1, kb11, st1);

    // row max across the 16 lanes that share each row (xor masks < 16)
    float rm[8], p0[8], p1[8], sc[8];
#pragma unroll
    for (int j = 0; j < 8; ++j) rm[j] = fmaxf(st0[j], st1[j]);
#pragma unroll
    for (int m = 1; m <= 8; m <<= 1)
#pragma unroll
      for (int j = 0; j < 8; ++j) rm[j] = fmaxf(rm[j], __shfl_xor(rm[j], m, 32));
#pragma unroll
    for (int j = 0; j < 8; ++j) {
      float mn = fmaxf(mrun[j], rm[j]);
      sc[j]   = __expf(mrun[j] - mn);
      mrun[j] = mn;
      p0[j] = __expf(st0[j] - mn);
      p1[j] = __expf(st1[j] - mn);
    }
    float rs[8];
#pragma unroll
    for (int j = 0; j < 8; ++j) rs[j] = p0[j] + p1[j];
#pragma unroll
    for (int m = 1; m <= 8; m <<= 1)
#pragma unroll
      for (int j = 0; j < 8; ++j) rs[j] += __shfl_xor(rs[j], m, 32);
#pragma unroll
    for (int j = 0; j < 8; ++j) lrun[j] = lrun[j] * sc[j] + rs[j];
#pragma unroll
    for (int c = 0; c < 4; ++c)
#pragma unroll
      for (int j = 0; j < 8; ++j) ctx[c][j] *= sc[j];

    // re-stripe P (C-layout) -> A-fragment layout via per-wave LDS buffer
#pragma unroll
    for (int j = 0; j < 8; ++j) {
      pw[(j + hi * 8) * 32 + cl]      = (bf16)p0[j];
      pw[(j + hi * 8) * 32 + 16 + cl] = (bf16)p1[j];
    }
    __asm__ volatile("" ::: "memory");
    v16bf pa;
    {
      const bf16* pr = pw + cl * 32 + hi * 16;
      v8bf a0 = *(const v8bf*)(pr);
      v8bf a1 = *(const v8bf*)(pr + 8);
#pragma unroll
      for (int i = 0; i < 8; ++i) { pa[i] = a0[i]; pa[i + 8] = a1[i]; }
    }
    __asm__ volatile("" ::: "memory");

    // ctx += P(16x32) * V(32x64); V^T rows in LDS are K(s)-contiguous
    const bf16* vc = &vsh[cur][0];
    ctx[0] = wmma_bf16(pa, *(const v16bf*)(vc + (0  + cl) * WP_ + hi * 16), ctx[0]);
    ctx[1] = wmma_bf16(pa, *(const v16bf*)(vc + (16 + cl) * WP_ + hi * 16), ctx[1]);
    ctx[2] = wmma_bf16(pa, *(const v16bf*)(vc + (32 + cl) * WP_ + hi * 16), ctx[2]);
    ctx[3] = wmma_bf16(pa, *(const v16bf*)(vc + (48 + cl) * WP_ + hi * 16), ctx[3]);

    wait_async0();
    __syncthreads();
  }

  float inv[8];
#pragma unroll
  for (int j = 0; j < 8; ++j) inv[j] = 1.0f / lrun[j];
#pragma unroll
  for (int c = 0; c < 4; ++c)
#pragma unroll
    for (int j = 0; j < 8; ++j) {
      const int l = l0 + j + hi * 8;
      const int d = 16 * c + cl;
      ctxb[((size_t)l * NB_ + n) * E_ + h * D_ + d] = (bf16)(ctx[c][j] * inv[j]);
    }
  if (cl == 0) {
#pragma unroll
    for (int j = 0; j < 8; ++j) {
      const int l = l0 + j + hi * 8;
      mstat[(size_t)nh * L_ + l] = mrun[j];
      lstat[(size_t)nh * L_ + l] = lrun[j];
    }
  }
}

// ---------------------------------------------------------------------------
// Kernel 3: attn_weights[n,l,s] = mean_h softmax(q k^T): recompute scores via
// WMMA, normalize with stored (m,lsum), average over the 16 heads. No atomics.
// ---------------------------------------------------------------------------
__global__ __launch_bounds__(128) void attn_weights_kernel(
    const bf16* __restrict__ qh, const bf16* __restrict__ kh,
    const float* __restrict__ mstat, const float* __restrict__ lstat,
    float* __restrict__ attn_out) {
  __shared__ float smax[4][16][16];  // [wave][row][head]
  __shared__ float sinv[4][16][16];
  const int lane = threadIdx.x & 31;
  const int w    = threadIdx.x >> 5;
  const int hi   = (lane >> 4) & 1;
  const int cl   = lane & 15;
  const int wid  = blockIdx.x * 4 + w;
  const int n    = wid >> 7;
  const int l0   = (wid & 127) << 4;

  for (int i = lane; i < 256; i += 32) {
    const int h = i >> 4, r = i & 15;
    const size_t idx = (size_t)(n * H_ + h) * L_ + l0 + r;
    smax[w][r][h] = mstat[idx];
    sinv[w][r][h] = 1.0f / lstat[idx];
  }
  __asm__ volatile("" ::: "memory");

  for (int s0 = 0; s0 < S_; s0 += 32) {
    float a0[8] = {}, a1[8] = {};
    for (int h = 0; h < H_; ++h) {
      const bf16* qrow =
          qh + (size_t)(n * H_ + h) * L_ * D_ + (size_t)(l0 + cl) * D_;
      const bf16* kbh = kh + (size_t)(n * H_ + h) * S_ * D_;
      v16bf qA0 = a_frag_bf16(qrow, 0, hi);
      v16bf qA1 = a_frag_bf16(qrow, 32, hi);
      const bf16* kc0 = kbh + (size_t)(s0 + cl) * D_;
      const bf16* kc1 = kbh + (size_t)(s0 + 16 + cl) * D_;
      v8f z = {};
      v8f st0 = wmma_bf16(qA0, *(const v16bf*)(kc0 + hi * 16), z);
      st0     = wmma_bf16(qA1, *(const v16bf*)(kc0 + 32 + hi * 16), st0);
      v8f st1 = wmma_bf16(qA0, *(const v16bf*)(kc1 + hi * 16), z);
      st1     = wmma_bf16(qA1, *(const v16bf*)(kc1 + 32 + hi * 16), st1);
#pragma unroll
      for (int j = 0; j < 8; ++j) {
        const float m  = smax[w][j + hi * 8][h];
        const float iv = sinv[w][j + hi * 8][h];
        a0[j] += __expf(st0[j] - m) * iv;
        a1[j] += __expf(st1[j] - m) * iv;
      }
    }
#pragma unroll
    for (int j = 0; j < 8; ++j) {
      const int l = l0 + j + hi * 8;
      float* dst = attn_out + ((size_t)n * L_ + l) * S_ + s0;
      dst[cl]      = a0[j] * 0.0625f;  // 1/H
      dst[16 + cl] = a1[j] * 0.0625f;
    }
  }
}

// ---------------------------------------------------------------------------
// Kernel 4: out projection. out[l,n,f] = sum_e ctx[l,n,e]*w[f,e] + b[f]
// Same LDS-staged structure as kernel 1 (bf16 A rows from global).
// ---------------------------------------------------------------------------
__global__ __launch_bounds__(128) void out_proj_kernel(
    const bf16* __restrict__ ctxb, const bf16* __restrict__ wbf,
    const float* __restrict__ opb, float* __restrict__ out) {
  __shared__ __align__(16) bf16 bsh[2][64 * WP_];
  const int lane = threadIdx.x & 31;
  const int w    = threadIdx.x >> 5;
  const int hi   = (lane >> 4) & 1;
  const int cl   = lane & 15;
  const int wid  = blockIdx.x * 4 + w;
  const int mt   = wid & 511;
  const int fc   = wid >> 9;  // 16 chunks of 64
  const int m0   = mt << 4;
  const int f0   = fc << 6;

  const bf16* arow   = ctxb + (size_t)(m0 + cl) * E_;
  const bf16* wpanel = wbf + (size_t)f0 * E_;

  stage_w(&bsh[0][0], wpanel, 0, w, lane);
  wait_async0();
  __syncthreads();

  v8f acc[4] = {};
  v16bf a_cur = a_frag_bf16(arow, 0, hi);
  for (int it = 0; it < E_ / 32; ++it) {
    const int k0  = it * 32;
    const int cur = it & 1;
    if (k0 + 32 < E_) stage_w(&bsh[cur ^ 1][0], wpanel, k0 + 32, w, lane);
    v16bf a_nxt = (k0 + 32 < E_) ? a_frag_bf16(arow, k0 + 32, hi) : a_cur;
    const bf16* bp = &bsh[cur][0];
    v16bf b0 = b_frag_lds(bp + (size_t)(0  + cl) * WP_, hi);
    v16bf b1 = b_frag_lds(bp + (size_t)(16 + cl) * WP_, hi);
    v16bf b2 = b_frag_lds(bp + (size_t)(32 + cl) * WP_, hi);
    v16bf b3 = b_frag_lds(bp + (size_t)(48 + cl) * WP_, hi);
    acc[0] = wmma_bf16(a_cur, b0, acc[0]);
    acc[1] = wmma_bf16(a_cur, b1, acc[1]);
    acc[2] = wmma_bf16(a_cur, b2, acc[2]);
    acc[3] = wmma_bf16(a_cur, b3, acc[3]);
    a_cur = a_nxt;
    wait_async0();
    __syncthreads();
  }

#pragma unroll
  for (int c = 0; c < 4; ++c) {
    const int f = f0 + 16 * c + cl;
    const float bias = opb[f];
#pragma unroll
    for (int j = 0; j < 8; ++j) {
      const int r = m0 + j + hi * 8;
      out[(size_t)r * E_ + f] = acc[c][j] + bias;
    }
  }
}

// ---------------------------------------------------------------------------
extern "C" void kernel_launch(void* const* d_in, const int* in_sizes, int n_in,
                              void* d_out, int out_size, void* d_ws,
                              size_t ws_size, hipStream_t stream) {
  const float* query = (const float*)d_in[0];
  const float* keyx  = (const float*)d_in[1];
  const float* value = (const float*)d_in[2];
  const float* ipw   = (const float*)d_in[3];
  const float* ipb   = (const float*)d_in[4];
  const float* opw   = (const float*)d_in[5];
  const float* opb   = (const float*)d_in[6];

  float* out  = (float*)d_out;
  float* attn = out + (size_t)L_ * NB_ * E_;  // attn_weights region

  constexpr size_t PROJ_ELEMS = (size_t)NB_ * H_ * L_ * D_;  // 8,388,608
  bf16* ws_q  = (bf16*)d_ws;
  bf16* ws_k  = ws_q + PROJ_ELEMS;
  bf16* ws_v  = ws_k + PROJ_ELEMS;               // transposed [n][h][d][s]
  bf16* ws_c  = ws_v + PROJ_ELEMS;               // ctx bf16 [l][n][E]
  bf16* ws_wi = ws_c + PROJ_ELEMS;               // in_proj_w bf16 (3E x E)
  bf16* ws_wo = ws_wi + (size_t)3 * E_ * E_;     // out_proj_w bf16 (E x E)
  float* ws_m = (float*)(ws_wo + (size_t)E_ * E_);
  float* ws_l = ws_m + (size_t)NB_ * H_ * L_;

  // K0: weight fp32 -> bf16
  convert_w_kernel<<<4096, 256, 0, stream>>>(ipw, opw, ws_wi, ws_wo);
  // K1: 512 m-tiles * 48 f-chunks = 24576 waves / 4 waves-per-block
  qkv_proj_kernel<<<6144, 128, 0, stream>>>(query, keyx, value, ws_wi, ipb,
                                            ws_q, ws_k, ws_v);
  // K2: 64 (n,h) * 128 l-tiles = 8192 waves
  flash_attn_kernel<<<2048, 128, 0, stream>>>(ws_q, ws_k, ws_v, ws_c, ws_m,
                                              ws_l);
  // K3: 4 n * 128 l-tiles = 512 waves
  attn_weights_kernel<<<128, 128, 0, stream>>>(ws_q, ws_k, ws_m, ws_l, attn);
  // K4: 512 m-tiles * 16 f-chunks = 8192 waves
  out_proj_kernel<<<2048, 128, 0, stream>>>(ws_c, ws_wo, opb, out);
}
